// Sub_lstm_35777077576479
// MI455X (gfx1250) — matrix operand order repaired and checked by reading, not compiled, not run on hardware
//
#include <hip/hip_runtime.h>
#include <cstddef>

// ---------------------------------------------------------------------------
// Problem constants (from reference): B=128, T=96, IN=64, H=64, 4H=256, 25 iters
// ---------------------------------------------------------------------------
#define BB    128
#define TT    96
#define INF_  64
#define HH    64
#define FOURH 256
#define NIT   25

typedef _Float16 v16h __attribute__((ext_vector_type(16)));
typedef _Float16 v8h  __attribute__((ext_vector_type(8)));
typedef float    v8f  __attribute__((ext_vector_type(8)));

__device__ __forceinline__ v8f wmma_f16(v16h a, v16h b, v8f c) {
  // D = A(16x32 f16) * B(32x16 f16) + C(16x16 f32)
  return __builtin_amdgcn_wmma_f32_16x16x32_f16(false, a, false, b, (short)0, c,
                                                false, false);
}

__device__ __forceinline__ float fast_sigmoid(float x) {
  return __builtin_amdgcn_rcpf(1.0f + __expf(-x));
}
__device__ __forceinline__ float fast_tanh(float x) {
  return 1.0f - 2.0f * __builtin_amdgcn_rcpf(__expf(2.0f * x) + 1.0f);
}

// ---------------------------------------------------------------------------
// Kernel 1: Xg[t][mt][nt] fragments = x[:,t,:] @ Wih^T + (bih+bhh)
// Stored fragment-major: idx = (((t*8+mt)*16+nt)*32 + lane)*8 + v   (f32)
// One wave per (t, mt) row-tile; 32 WMMAs per wave.
// ---------------------------------------------------------------------------
__global__ __launch_bounds__(256, 1) void xgemm_kernel(
    const float* __restrict__ x, const float* __restrict__ Wih,
    const float* __restrict__ bih, const float* __restrict__ bhh,
    float* __restrict__ Xg) {
  const int t    = blockIdx.x;
  const int lane = threadIdx.x & 31;
  const int mt   = threadIdx.x >> 5;      // 0..7 : batch row tile
  const int g    = lane >> 4;
  const int r    = lane & 15;

  // B fragments of Wih^T : B[k][n] = Wih[n*IN + k]
  v16h Bf[2][16];
#pragma unroll
  for (int kt = 0; kt < 2; ++kt)
#pragma unroll
    for (int nt = 0; nt < 16; ++nt) {
      const float* p = Wih + (nt * 16 + r) * INF_ + kt * 32 + 16 * g;
#pragma unroll
      for (int j = 0; j < 16; ++j) Bf[kt][nt][j] = (_Float16)p[j];
    }

  float bbv[16];
#pragma unroll
  for (int nt = 0; nt < 16; ++nt) {
    const int n = nt * 16 + r;
    bbv[nt] = bih[n] + bhh[n];
  }

  // A fragments from x rows (mt*16 + r), time slice t
  v16h A[2];
  const float* xr = x + (size_t)(mt * 16 + r) * (TT * INF_) + t * INF_;
#pragma unroll
  for (int kt = 0; kt < 2; ++kt) {
    const float* p = xr + kt * 32;
#pragma unroll
    for (int j = 0; j < 8; ++j) A[kt][j] = (_Float16)p[16 * g + j];
#pragma unroll
    for (int j = 0; j < 8; ++j) A[kt][j + 8] = (_Float16)p[16 * g + 8 + j];
  }

  float* og = Xg + (size_t)((t * 8 + mt) * 16) * 256;  // + nt*256 + lane*8
#pragma unroll
  for (int nt = 0; nt < 16; ++nt) {
    v8f acc;
#pragma unroll
    for (int v = 0; v < 8; ++v) acc[v] = bbv[nt];
    acc = wmma_f16(A[0], Bf[0][nt], acc);
    acc = wmma_f16(A[1], Bf[1][nt], acc);
    *(v8f*)(og + nt * 256 + lane * 8) = acc;
  }
}

// ---------------------------------------------------------------------------
// Kernel 2: 27 independent LSTM scans x 8 batch-row tiles = 216 single-wave
// blocks. Whh^T lives in 256 VGPRs as 32 B-fragments for the whole kernel.
// Per step: LDS h -> A frags (ds_load_b128), z = Xg frag + h@Whh^T (WMMA),
// fused i/f/g/o gates elementwise on the D-fragment layout, h back to LDS.
//   job  0..24 : suffix scan over t = job+1 .. 95, write final h -> hsuf[job]
//   job  25    : prefix scan  t = 0..24, write h each step -> hpref[s]
//   job  26    : reverse scan t = 95..71, write h each step -> hfpref[s]
// ---------------------------------------------------------------------------
__global__ __launch_bounds__(32, 1) void lstm_scan_kernel(
    const float* __restrict__ Whh, const float* __restrict__ Xg,
    float* __restrict__ hpref, float* __restrict__ hfpref,
    float* __restrict__ hsuf) {
  const int job  = blockIdx.x >> 3;   // 0..26
  const int mt   = blockIdx.x & 7;    // batch row tile
  const int lane = threadIdx.x & 31;
  const int g    = lane >> 4;
  const int r    = lane & 15;

  // B fragments of Whh^T : B[k][n] = Whh[n*H + k]  (register resident)
  v16h Bf[2][16];
#pragma unroll
  for (int kt = 0; kt < 2; ++kt)
#pragma unroll
    for (int nt = 0; nt < 16; ++nt) {
      const float* p = Whh + (nt * 16 + r) * HH + kt * 32 + 16 * g;
#pragma unroll
      for (int j = 0; j < 16; ++j) Bf[kt][nt][j] = (_Float16)p[j];
    }

  // per-wave h strip (16 rows x 64 cols, f16) for D-layout -> A-layout relayout
  __shared__ alignas(16) _Float16 hsh[16 * 64];
  for (int idx = lane; idx < 16 * 64; idx += 32) hsh[idx] = (_Float16)0.0f;

  v8f c[4];
#pragma unroll
  for (int ct = 0; ct < 4; ++ct)
#pragma unroll
    for (int v = 0; v < 8; ++v) c[ct][v] = 0.0f;

  int t0, tstep, nsteps, mode;
  if (job < NIT)        { t0 = job + 1; tstep = 1;  nsteps = TT - (job + 1); mode = 0; }
  else if (job == NIT)  { t0 = 0;       tstep = 1;  nsteps = NIT;            mode = 1; }
  else                  { t0 = TT - 1;  tstep = -1; nsteps = NIT;            mode = 2; }

  int t = t0;
  for (int s = 0; s < nsteps; ++s, t += tstep) {
    // ---- A fragments (this wave's own h rows) from LDS -------------------
    v16h A[2];
#pragma unroll
    for (int kt = 0; kt < 2; ++kt) {
      v8h lo = *(const v8h*)&hsh[r * 64 + kt * 32 + 16 * g];
      v8h hi = *(const v8h*)&hsh[r * 64 + kt * 32 + 16 * g + 8];
#pragma unroll
      for (int j = 0; j < 8; ++j) { A[kt][j] = lo[j]; A[kt][j + 8] = hi[j]; }
    }

    float* outp = nullptr;
    if (mode == 1)                 outp = hpref  + (size_t)s   * (BB * HH);
    else if (mode == 2)            outp = hfpref + (size_t)s   * (BB * HH);
    else if (s == nsteps - 1)      outp = hsuf   + (size_t)job * (BB * HH);

    const float* xg = Xg + (size_t)((t * 8 + mt) * 16) * 256 + lane * 8;

#pragma unroll
    for (int ct = 0; ct < 4; ++ct) {
      v8f zi = *(const v8f*)(xg + (ct     ) * 256);
      v8f zf = *(const v8f*)(xg + (ct +  4) * 256);
      v8f zg = *(const v8f*)(xg + (ct +  8) * 256);
      v8f zo = *(const v8f*)(xg + (ct + 12) * 256);
      zi = wmma_f16(A[0], Bf[0][ct     ], zi); zi = wmma_f16(A[1], Bf[1][ct     ], zi);
      zf = wmma_f16(A[0], Bf[0][ct +  4], zf); zf = wmma_f16(A[1], Bf[1][ct +  4], zf);
      zg = wmma_f16(A[0], Bf[0][ct +  8], zg); zg = wmma_f16(A[1], Bf[1][ct +  8], zg);
      zo = wmma_f16(A[0], Bf[0][ct + 12], zo); zo = wmma_f16(A[1], Bf[1][ct + 12], zo);
#pragma unroll
      for (int v = 0; v < 8; ++v) {
        const float ig = fast_sigmoid(zi[v]);
        const float fg = fast_sigmoid(zf[v]);
        const float gg = fast_tanh(zg[v]);
        const float og = fast_sigmoid(zo[v]);
        const float cn = fg * c[ct][v] + ig * gg;
        c[ct][v] = cn;
        const float hn = og * fast_tanh(cn);
        hsh[(v + 8 * g) * 64 + ct * 16 + r] = (_Float16)hn;
        if (outp) outp[(mt * 16 + v + 8 * g) * HH + ct * 16 + r] = hn;
      }
    }
  }
}

// ---------------------------------------------------------------------------
// Kernel 3: Hs[b][ch] = max over 25 iterations of the concatenated maxpools.
// ---------------------------------------------------------------------------
__global__ void hs_max_kernel(const float* __restrict__ hpref,
                              const float* __restrict__ hfpref,
                              const float* __restrict__ hsuf,
                              float* __restrict__ Hs) {
  const int idx = blockIdx.x * blockDim.x + threadIdx.x;  // 128*128
  const int b = idx >> 7;
  const int ch = idx & 127;
  float m = -3.402823466e38f;
  if (ch < HH) {
#pragma unroll 5
    for (int i = 0; i < NIT; ++i) {
      const float a = hpref[i * (BB * HH) + b * HH + ch];
      const float s = hsuf [i * (BB * HH) + b * HH + ch];
      m = fmaxf(m, fmaxf(a, s));
    }
  } else {
    const int c2 = ch - HH;
#pragma unroll 5
    for (int i = 0; i < NIT; ++i) {
      const float s = hsuf  [i * (BB * HH) + b * HH + c2];
      const float f = hfpref[i * (BB * HH) + b * HH + c2];
      m = fmaxf(m, fmaxf(s, f));
    }
  }
  Hs[idx] = m;
}

// ---------------------------------------------------------------------------
// Kernel 4: out[b][n][ch] = Hs[b][ch] * Wlin[n] + blin[n]   (B, 26, 2H)
// ---------------------------------------------------------------------------
__global__ void linear_kernel(const float* __restrict__ Hs,
                              const float* __restrict__ Wlin,
                              const float* __restrict__ blin,
                              float* __restrict__ out, int total) {
  const int idx = blockIdx.x * blockDim.x + threadIdx.x;
  if (idx >= total) return;
  const int ch = idx & 127;
  const int n  = (idx >> 7) % 26;
  const int b  = idx / (26 * 128);
  out[idx] = Hs[b * 128 + ch] * Wlin[n] + blin[n];
}

// ---------------------------------------------------------------------------
extern "C" void kernel_launch(void* const* d_in, const int* in_sizes, int n_in,
                              void* d_out, int out_size, void* d_ws,
                              size_t ws_size, hipStream_t stream) {
  const float* x    = (const float*)d_in[0];
  const float* Wih  = (const float*)d_in[1];
  const float* Whh  = (const float*)d_in[2];
  const float* bih  = (const float*)d_in[3];
  const float* bhh  = (const float*)d_in[4];
  const float* Wlin = (const float*)d_in[5];
  const float* blin = (const float*)d_in[6];
  float* out = (float*)d_out;
  float* ws  = (float*)d_ws;

  // workspace layout (floats):
  float* Xg     = ws;                  // 96*8*16*32*8 = 3,145,728
  float* hpref  = Xg     + 3145728;    // 25*128*64    =   204,800
  float* hfpref = hpref  + 204800;     //                  204,800
  float* hsuf   = hfpref + 204800;     //                  204,800
  float* Hs     = hsuf   + 204800;     // 128*128      =    16,384
  (void)ws_size; (void)in_sizes; (void)n_in; (void)out_size;

  hipLaunchKernelGGL(xgemm_kernel, dim3(96), dim3(256), 0, stream,
                     x, Wih, bih, bhh, Xg);
  hipLaunchKernelGGL(lstm_scan_kernel, dim3(216), dim3(32), 0, stream,
                     Whh, Xg, hpref, hfpref, hsuf);
  hipLaunchKernelGGL(hs_max_kernel, dim3(64), dim3(256), 0, stream,
                     hpref, hfpref, hsuf, Hs);
  hipLaunchKernelGGL(linear_kernel, dim3(1664), dim3(256), 0, stream,
                     Hs, Wlin, blin, out, BB * 26 * 128);
}
